// BidirRWKV6TimeMix_76003741270340
// MI455X (gfx1250) — matrix operand order repaired and checked by reading, not compiled, not run on hardware
//
#include <hip/hip_runtime.h>
#include <hip/hip_bf16.h>
#include <math.h>

typedef __bf16 bf16;
typedef __attribute__((ext_vector_type(16))) __bf16 v16bf;
typedef __attribute__((ext_vector_type(8)))  __bf16 v8bf;
typedef __attribute__((ext_vector_type(8)))  float  v8f;

#define DD   1024
#define HH   16
#define KK   64
#define BBn  4
#define TT   1024
#define DMIX 32
#define BT   (BBn*TT)                 // 4096
#define NELEM ((size_t)BBn*TT*DD)     // 4,194,304

// ---------------- WMMA fragment helpers (CDNA5 wave32 layouts) ----------------
__device__ __forceinline__ v16bf cat16(v8bf lo, v8bf hi) {
  return __builtin_shufflevector(lo, hi, 0,1,2,3,4,5,6,7,8,9,10,11,12,13,14,15);
}
// A (16xK=32, bf16): lane l = row M; lanes<16 hold K 0-15, lanes>=16 hold K 16-31 (contiguous per lane)
__device__ __forceinline__ v16bf frag_a(const bf16* base, int ld, int k0, int lane) {
  int l = lane & 15, hi = (lane >> 4) & 1;
  const bf16* p = base + (size_t)l * ld + k0 + hi * 16;
  return cat16(*(const v8bf*)p, *(const v8bf*)(p + 8));
}
// B (K=32 x 16, bf16) accessed as rows of a (N x K) row-major matrix:
// lane l = column N; lanes<16 hold K 0-15, lanes>=16 hold K 16-31 (contiguous per lane)
__device__ __forceinline__ v16bf frag_b(const bf16* base, int ld, int k0, int lane) {
  int l = lane & 15, hi = (lane >> 4) & 1;
  const bf16* p = base + (size_t)l * ld + k0 + hi * 16;
  return cat16(*(const v8bf*)p, *(const v8bf*)(p + 8));
}
__device__ __forceinline__ v8f wmma_bf(v16bf a, v16bf b, v8f c) {
  return __builtin_amdgcn_wmma_f32_16x16x32_bf16(false, a, false, b, (short)0, c, false, false);
}

struct Frag4 { v16bf f[4]; };

// 4 k-fragments covering a 32(j) x 64(feature) K-block: (js=0..1) x (k0=0,32)
__device__ __forceinline__ Frag4 load_k4(const bf16* kB, int j0, int lane) {
  Frag4 r;
  r.f[0] = frag_b(kB + (size_t)j0 * DD, DD, 0, lane);
  r.f[1] = frag_b(kB + (size_t)j0 * DD, DD, 32, lane);
  r.f[2] = frag_b(kB + (size_t)(j0 + 16) * DD, DD, 0, lane);
  r.f[3] = frag_b(kB + (size_t)(j0 + 16) * DD, DD, 32, lane);
  return r;
}
// 4 v^T fragments: feature tiles 0..3, K = j-chunk of 32
__device__ __forceinline__ Frag4 load_v4(const bf16* vTB, int j0, int lane) {
  Frag4 r;
#pragma unroll
  for (int f = 0; f < 4; ++f)
    r.f[f] = frag_b(vTB + (size_t)(f * 16) * TT, TT, j0, lane);
  return r;
}
// two 16x16 A-subtiles (fp32) for one 32-wide j-chunk
__device__ __forceinline__ void atiles(v16bf A0, v16bf A1, const Frag4& k4, v8f at[2]) {
  v8f z = {};
  at[0] = wmma_bf(A0, k4.f[0], z);
  at[0] = wmma_bf(A1, k4.f[1], at[0]);
  at[1] = wmma_bf(A0, k4.f[2], z);
  at[1] = wmma_bf(A1, k4.f[3], at[1]);
}
// C-layout fp32 tiles -> bf16 LDS (row-major 16x32) -> A-fragment -> 4 y WMMAs
__device__ __forceinline__ void stage_and_acc(bf16* st, const v8f at[2], const Frag4& vb,
                                              v8f acc[4], int lane) {
  int l = lane & 15, hi = (lane >> 4) & 1;
#pragma unroll
  for (int js = 0; js < 2; ++js)
#pragma unroll
    for (int vg = 0; vg < 8; ++vg)
      st[(hi * 8 + vg) * 32 + js * 16 + l] = (bf16)at[js][vg];
  v16bf aa = frag_a(st, 32, 0, lane);  // same-wave LDS ops are in-order
#pragma unroll
  for (int f = 0; f < 4; ++f)
    acc[f] = wmma_bf(aa, vb.f[f], acc[f]);
}

// ---------------- weight conversion ----------------
__global__ void cvt_bf16_k(const float* __restrict__ s, bf16* __restrict__ d, int n) {
  int i = blockIdx.x * blockDim.x + threadIdx.x;
  if (i < n) d[i] = (bf16)s[i];
}
// dst (C x R) row-major = transpose of src (R x C)
__global__ void cvtT_bf16_k(const float* __restrict__ s, bf16* __restrict__ d, int R, int C) {
  int o = blockIdx.x * blockDim.x + threadIdx.x;
  if (o >= R * C) return;
  int c = o / R, r = o - c * R;
  d[o] = (bf16)s[(size_t)r * C + c];
}

// ---------------- stage 1: bidir shift + mix input ----------------
__global__ void prep_k(const float* __restrict__ x, const float* __restrict__ maa_x,
                       float* __restrict__ dxprev, bf16* __restrict__ xmix) {
  size_t idx = (size_t)blockIdx.x * blockDim.x + threadIdx.x;
  if (idx >= NELEM) return;
  int d = idx % DD;
  int t = (idx / DD) % TT;
  float xc = x[idx];
  float xl = (t > 0)      ? x[idx - DD] : 0.0f;
  float xr = (t < TT - 1) ? x[idx + DD] : 0.0f;
  float dx = 0.5f * (xl + xr) - xc;
  dxprev[idx] = dx;
  xmix[idx] = (bf16)(xc + dx * maa_x[d]);
}

// ---------------- generic pipelined WMMA GEMM: C = A(MxK) * Bt(NxK)^T ----------------
// wave tile: 32(M) x 64(N), double-buffered fragments over K.
// EPI: 0 = f32 store, 1 = tanh->f32, 3 = tanh->bf16, 4 = -exp(bias[n]+acc)->f32
template <int EPI>
__global__ __launch_bounds__(256) void gemm_bt(const bf16* __restrict__ A, const bf16* __restrict__ Bt,
                                               float* __restrict__ C, bf16* __restrict__ Cb,
                                               const float* __restrict__ bias, int M, int N, int K) {
  int lane = threadIdx.x & 31;
  int wave = threadIdx.x >> 5;
  int m0 = blockIdx.x * 256 + wave * 32;
  int n0 = blockIdx.y * 64;
  if (m0 >= M) return;
  int l = lane & 15, hi = (lane >> 4) & 1;

  const bf16* A0 = A + (size_t)m0 * K;
  const bf16* A1 = A0 + (size_t)16 * K;
  // clamped B row pointers (per n-tile), loads OOB-safe; stores guarded
  const bf16* Bp[4];
#pragma unroll
  for (int t = 0; t < 4; ++t) {
    int n = n0 + t * 16 + l;
    n = (n < N) ? n : (N - 1);
    Bp[t] = Bt + (size_t)n * K + hi * 16;
  }

  v8f acc[8] = {};
  // prologue: fragments for k0 = 0
  v16bf a0 = frag_a(A0, K, 0, lane);
  v16bf a1 = frag_a(A1, K, 0, lane);
  v16bf b[4];
#pragma unroll
  for (int t = 0; t < 4; ++t)
    b[t] = cat16(*(const v8bf*)(Bp[t]), *(const v8bf*)(Bp[t] + 8));

  // pipelined main loop: load k0+32 while computing k0
  for (int k0 = 0; k0 + 32 < K; k0 += 32) {
    v16bf a0n = frag_a(A0, K, k0 + 32, lane);
    v16bf a1n = frag_a(A1, K, k0 + 32, lane);
    v16bf bn[4];
#pragma unroll
    for (int t = 0; t < 4; ++t)
      bn[t] = cat16(*(const v8bf*)(Bp[t] + k0 + 32), *(const v8bf*)(Bp[t] + k0 + 48));
#pragma unroll
    for (int t = 0; t < 4; ++t) {
      acc[t]     = wmma_bf(a0, b[t], acc[t]);
      acc[4 + t] = wmma_bf(a1, b[t], acc[4 + t]);
    }
    a0 = a0n; a1 = a1n;
#pragma unroll
    for (int t = 0; t < 4; ++t) b[t] = bn[t];
  }
  // epilogue: last k-step
#pragma unroll
  for (int t = 0; t < 4; ++t) {
    acc[t]     = wmma_bf(a0, b[t], acc[t]);
    acc[4 + t] = wmma_bf(a1, b[t], acc[4 + t]);
  }

#pragma unroll
  for (int sub = 0; sub < 2; ++sub)
#pragma unroll
    for (int t = 0; t < 4; ++t) {
      int n = n0 + t * 16 + l;
      if (n >= N) continue;
#pragma unroll
      for (int vg = 0; vg < 8; ++vg) {
        int m = m0 + sub * 16 + hi * 8 + vg;
        size_t idx = (size_t)m * N + n;
        float v = acc[sub * 4 + t][vg];
        if (EPI == 0) C[idx] = v;
        else if (EPI == 1) C[idx] = tanhf(v);
        else if (EPI == 3) Cb[idx] = (bf16)tanhf(v);
        else if (EPI == 4) C[idx] = -expf(bias[n] + v);
      }
    }
}

// ---------------- stage 2: 5-way low-rank mix (K=32 VALU dot) ----------------
__global__ void mix5_k(const float* __restrict__ x, const float* __restrict__ dxprev,
                       const float* __restrict__ xxx, const float* __restrict__ w2,
                       const float* __restrict__ mW, const float* __restrict__ mK,
                       const float* __restrict__ mV, const float* __restrict__ mR,
                       const float* __restrict__ mG,
                       bf16* __restrict__ xw, bf16* __restrict__ xk, bf16* __restrict__ xv,
                       bf16* __restrict__ xr, bf16* __restrict__ xg) {
  size_t idx = (size_t)blockIdx.x * blockDim.x + threadIdx.x;
  if (idx >= NELEM) return;
  int d = idx % DD;
  size_t bt = idx / DD;
  const float* xrow = xxx + bt * (5 * DMIX);
  float m[5];
#pragma unroll
  for (int f = 0; f < 5; ++f) {
    float s = 0.0f;
    for (int j = 0; j < DMIX; ++j)
      s += xrow[f * DMIX + j] * w2[((size_t)(f * DMIX + j)) * DD + d];
    m[f] = s;
  }
  float xc = x[idx], dx = dxprev[idx];
  xw[idx] = (bf16)(xc + dx * (mW[d] + m[0]));
  xk[idx] = (bf16)(xc + dx * (mK[d] + m[1]));
  xv[idx] = (bf16)(xc + dx * (mV[d] + m[2]));
  xr[idx] = (bf16)(xc + dx * (mR[d] + m[3]));
  xg[idx] = (bf16)(xc + dx * (mG[d] + m[4]));
}

// ---------------- stage 3: cumsum of w over T, capture midpoint shifts ----------------
__global__ void cumsum_k(const float* __restrict__ wneg, float* __restrict__ cs,
                         float* __restrict__ shF, float* __restrict__ shB) {
  int d = blockIdx.x * blockDim.x + threadIdx.x;
  int b = blockIdx.y;
  size_t base = (size_t)b * TT * DD + d;
  float run = 0.0f;
  for (int t = 0; t < TT; ++t) {
    float wv = wneg[base + (size_t)t * DD];
    run += wv;
    cs[base + (size_t)t * DD] = run;
    if (t == TT / 2) { shF[b * DD + d] = run; shB[b * DD + d] = run - wv; }
  }
}

// ---------------- stage 4: decay-weighted r/k factors (bf16) ----------------
__global__ void factors_k(const float* __restrict__ r32, const float* __restrict__ k32,
                          const float* __restrict__ cs, const float* __restrict__ wneg,
                          const float* __restrict__ shF, const float* __restrict__ shB,
                          bf16* __restrict__ rf, bf16* __restrict__ kf,
                          bf16* __restrict__ rb, bf16* __restrict__ kb) {
  size_t idx = (size_t)blockIdx.x * blockDim.x + threadIdx.x;
  if (idx >= NELEM) return;
  int d = idx % DD;
  int b = idx / ((size_t)TT * DD);
  float c  = cs[idx];
  float cf = fminf(fmaxf(c - shF[b * DD + d], -60.0f), 60.0f);
  float cb = fminf(fmaxf((c - wneg[idx]) - shB[b * DD + d], -60.0f), 60.0f);
  float rv = r32[idx], kv = k32[idx];
  rf[idx] = (bf16)(rv * expf(cf));
  kf[idx] = (bf16)(kv * expf(-cf));
  rb[idx] = (bf16)(rv * expf(-cb));
  kb[idx] = (bf16)(kv * expf(cb));
}

// ---------------- stage 5: v -> vT (B,H,K,T) bf16 ----------------
__global__ void vtrans_k(const float* __restrict__ v32, bf16* __restrict__ vT) {
  size_t idx = (size_t)blockIdx.x * blockDim.x + threadIdx.x;
  if (idx >= NELEM) return;
  int t = idx % TT;
  int f = (idx / TT) % KK;
  int h = (idx / ((size_t)TT * KK)) % HH;
  int b = idx / ((size_t)TT * KK * HH);
  vT[idx] = (bf16)v32[((size_t)b * TT + t) * DD + h * KK + f];
}

// ---------------- stage 6: bidirectional lion attention ----------------
// One wave per workgroup: i0 = blockIdx.x*16 is scalar -> all region control flow
// is s_cbranch (no exec masking); each uniform phase is software-pipelined.
__global__ __launch_bounds__(32) void lion_attn_k(const bf16* __restrict__ rf, const bf16* __restrict__ kf,
                                                  const bf16* __restrict__ rb, const bf16* __restrict__ kb,
                                                  const bf16* __restrict__ vT, float* __restrict__ y) {
  int lane = threadIdx.x & 31;
  int l = lane & 15, hi = (lane >> 4) & 1;
  int bh = blockIdx.y;
  int b = bh / HH, h = bh % HH;
  int i0 = blockIdx.x * 16;

  __shared__ __align__(16) bf16 stage[16][32];
  bf16* st = &stage[0][0];

  const bf16* rfB = rf + ((size_t)b * TT + i0) * DD + h * KK;
  const bf16* rbB = rb + ((size_t)b * TT + i0) * DD + h * KK;
  const bf16* kfB = kf + (size_t)b * TT * DD + h * KK;
  const bf16* kbB = kb + (size_t)b * TT * DD + h * KK;
  const bf16* vTB = vT + (size_t)bh * KK * TT;

  // loop-invariant row fragments
  v16bf af0 = frag_a(rfB, DD, 0, lane);
  v16bf af1 = frag_a(rfB, DD, 32, lane);
  v16bf ab0 = frag_a(rbB, DD, 0, lane);
  v16bf ab1 = frag_a(rbB, DD, 32, lane);

  v8f acc[4] = {};
  const int jMix = i0 & ~31;   // the single 32-wide chunk straddling the diagonal

  // ---- forward phase: j0 in [0, jMix), pipelined ----
  if (jMix > 0) {
    Frag4 kc = load_k4(kfB, 0, lane);
    Frag4 vc = load_v4(vTB, 0, lane);
    for (int j0 = 0; j0 < jMix; j0 += 32) {
      Frag4 kn = kc, vn = vc;
      if (j0 + 32 < jMix) { kn = load_k4(kfB, j0 + 32, lane); vn = load_v4(vTB, j0 + 32, lane); }
      v8f at[2];
      atiles(af0, af1, kc, at);
      stage_and_acc(st, at, vc, acc, lane);
      kc = kn; vc = vn;
    }
  }

  // ---- mixed step at jMix: both directions + per-element triangular select ----
  {
    int j0 = jMix;
    Frag4 kF = load_k4(kfB, j0, lane);
    Frag4 kB = load_k4(kbB, j0, lane);
    Frag4 vc = load_v4(vTB, j0, lane);
    v8f atF[2], atB[2];
    atiles(af0, af1, kF, atF);
    atiles(ab0, ab1, kB, atB);
#pragma unroll
    for (int js = 0; js < 2; ++js)
#pragma unroll
      for (int vg = 0; vg < 8; ++vg) {
        int ig = i0 + hi * 8 + vg;
        int jg = j0 + js * 16 + l;
        if (ig < jg) atF[js][vg] = atB[js][vg];
      }
    stage_and_acc(st, atF, vc, acc, lane);
  }

  // ---- backward phase: j0 in (jMix, TT), pipelined ----
  if (jMix + 32 < TT) {
    Frag4 kc = load_k4(kbB, jMix + 32, lane);
    Frag4 vc = load_v4(vTB, jMix + 32, lane);
    for (int j0 = jMix + 32; j0 < TT; j0 += 32) {
      Frag4 kn = kc, vn = vc;
      if (j0 + 32 < TT) { kn = load_k4(kbB, j0 + 32, lane); vn = load_v4(vTB, j0 + 32, lane); }
      v8f at[2];
      atiles(ab0, ab1, kc, at);
      stage_and_acc(st, at, vc, acc, lane);
      kc = kn; vc = vn;
    }
  }

#pragma unroll
  for (int f = 0; f < 4; ++f)
#pragma unroll
    for (int vg = 0; vg < 8; ++vg) {
      int t = i0 + hi * 8 + vg;
      y[((size_t)b * TT + t) * DD + h * KK + f * 16 + l] = acc[f][vg];
    }
}

// ---------------- stage 7: per-head groupnorm + silu gate ----------------
__global__ __launch_bounds__(256) void gn_gate_k(const float* __restrict__ y, const float* __restrict__ gpre,
                                                 const float* __restrict__ lnw, const float* __restrict__ lnb,
                                                 bf16* __restrict__ afin) {
  int lane = threadIdx.x & 31;
  int wave = threadIdx.x >> 5;
  int grp = blockIdx.x * 8 + wave;   // bt*HH + h
  int bt = grp >> 4;
  int h  = grp & 15;
  size_t base = (size_t)bt * DD + h * KK;
  float v0 = y[base + lane];
  float v1 = y[base + 32 + lane];
  float s = v0 + v1, s2 = v0 * v0 + v1 * v1;
  for (int off = 16; off > 0; off >>= 1) {
    s  += __shfl_xor(s, off);
    s2 += __shfl_xor(s2, off);
  }
  float mu  = s * (1.0f / 64.0f);
  float var = s2 * (1.0f / 64.0f) - mu * mu;
  float rstd = rsqrtf(var + 6.4e-4f);
  int d0 = h * KK + lane, d1 = d0 + 32;
  float g0 = gpre[base + lane],      sg0 = g0 / (1.0f + expf(-g0));
  float g1 = gpre[base + 32 + lane], sg1 = g1 / (1.0f + expf(-g1));
  afin[base + lane]      = (bf16)(((v0 - mu) * rstd * lnw[d0] + lnb[d0]) * sg0);
  afin[base + 32 + lane] = (bf16)(((v1 - mu) * rstd * lnw[d1] + lnb[d1]) * sg1);
}

// ---------------- host launch ----------------
extern "C" void kernel_launch(void* const* d_in, const int* in_sizes, int n_in,
                              void* d_out, int out_size, void* d_ws, size_t ws_size,
                              hipStream_t stream) {
  const float* x      = (const float*)d_in[0];
  const float* maa_x  = (const float*)d_in[1];
  const float* maa_w  = (const float*)d_in[2];
  const float* maa_k  = (const float*)d_in[3];
  const float* maa_v  = (const float*)d_in[4];
  const float* maa_r  = (const float*)d_in[5];
  const float* maa_g  = (const float*)d_in[6];
  const float* w1     = (const float*)d_in[7];   // (1024,160)
  const float* w2     = (const float*)d_in[8];   // (5,32,1024)
  const float* tdec   = (const float*)d_in[9];   // (1024)
  const float* dw1    = (const float*)d_in[10];  // (1024,64)
  const float* dw2    = (const float*)d_in[11];  // (64,1024)
  const float* Wr     = (const float*)d_in[12];
  const float* Wk     = (const float*)d_in[13];
  const float* Wv     = (const float*)d_in[14];
  const float* Wg     = (const float*)d_in[15];
  const float* Wo     = (const float*)d_in[16];
  const float* lnw    = (const float*)d_in[17];
  const float* lnb    = (const float*)d_in[18];
  float* out = (float*)d_out;

  char* p = (char*)d_ws;
  auto alloc = [&](size_t bytes) -> void* {
    void* r = (void*)p;
    p += (bytes + 255) & ~(size_t)255;
    return r;
  };
  float* dxprev = (float*)alloc(NELEM * 4);
  bf16*  xmix   = (bf16*) alloc(NELEM * 2);
  bf16*  w1T    = (bf16*) alloc((size_t)160 * DD * 2);
  bf16*  dw1T   = (bf16*) alloc((size_t)64 * DD * 2);
  bf16*  dw2T   = (bf16*) alloc((size_t)DD * 64 * 2);
  bf16*  Wrb    = (bf16*) alloc((size_t)DD * DD * 2);
  bf16*  Wkb    = (bf16*) alloc((size_t)DD * DD * 2);
  bf16*  Wvb    = (bf16*) alloc((size_t)DD * DD * 2);
  bf16*  Wgb    = (bf16*) alloc((size_t)DD * DD * 2);
  bf16*  Wob    = (bf16*) alloc((size_t)DD * DD * 2);
  float* xxx    = (float*)alloc((size_t)BT * 160 * 4);
  bf16*  xw     = (bf16*) alloc(NELEM * 2);
  bf16*  xk     = (bf16*) alloc(NELEM * 2);
  bf16*  xv     = (bf16*) alloc(NELEM * 2);
  bf16*  xr     = (bf16*) alloc(NELEM * 2);
  bf16*  xg     = (bf16*) alloc(NELEM * 2);
  float* r32    = (float*)alloc(NELEM * 4);
  float* k32    = (float*)alloc(NELEM * 4);
  float* v32    = (float*)alloc(NELEM * 4);
  float* gpre   = (float*)alloc(NELEM * 4);
  bf16*  tmpd   = (bf16*) alloc((size_t)BT * 64 * 2);
  float* wneg   = (float*)alloc(NELEM * 4);
  float* cs     = (float*)alloc(NELEM * 4);
  float* shF    = (float*)alloc((size_t)BBn * DD * 4);
  float* shB    = (float*)alloc((size_t)BBn * DD * 4);
  bf16*  rff    = (bf16*) alloc(NELEM * 2);
  bf16*  kff    = (bf16*) alloc(NELEM * 2);
  bf16*  rbb    = (bf16*) alloc(NELEM * 2);
  bf16*  kbb    = (bf16*) alloc(NELEM * 2);
  bf16*  vT     = (bf16*) alloc(NELEM * 2);
  float* yb     = (float*)alloc(NELEM * 4);
  bf16*  afin   = (bf16*) alloc(NELEM * 2);

  const int TPB = 256;
  int gElem = (int)((NELEM + TPB - 1) / TPB);
  auto ggrid = [&](int M, int N) { return dim3((M + 255) / 256, (N + 63) / 64); };

  // weight conversion
  cvtT_bf16_k<<<(160 * DD + TPB - 1) / TPB, TPB, 0, stream>>>(w1, w1T, DD, 160);
  cvtT_bf16_k<<<(64 * DD + TPB - 1) / TPB, TPB, 0, stream>>>(dw1, dw1T, DD, 64);
  cvtT_bf16_k<<<(DD * 64 + TPB - 1) / TPB, TPB, 0, stream>>>(dw2, dw2T, 64, DD);
  int nW = DD * DD;
  cvt_bf16_k<<<(nW + TPB - 1) / TPB, TPB, 0, stream>>>(Wr, Wrb, nW);
  cvt_bf16_k<<<(nW + TPB - 1) / TPB, TPB, 0, stream>>>(Wk, Wkb, nW);
  cvt_bf16_k<<<(nW + TPB - 1) / TPB, TPB, 0, stream>>>(Wv, Wvb, nW);
  cvt_bf16_k<<<(nW + TPB - 1) / TPB, TPB, 0, stream>>>(Wg, Wgb, nW);
  cvt_bf16_k<<<(nW + TPB - 1) / TPB, TPB, 0, stream>>>(Wo, Wob, nW);

  // token shift + mix input
  prep_k<<<gElem, TPB, 0, stream>>>(x, maa_x, dxprev, xmix);

  // xxx = tanh(xmix @ w1)
  gemm_bt<1><<<ggrid(BT, 160), TPB, 0, stream>>>(xmix, w1T, xxx, nullptr, nullptr, BT, 160, DD);

  // 5-way mix -> xw..xg
  mix5_k<<<gElem, TPB, 0, stream>>>(x, dxprev, xxx, w2, maa_w, maa_k, maa_v, maa_r, maa_g,
                                    xw, xk, xv, xr, xg);

  // projections
  gemm_bt<0><<<ggrid(BT, DD), TPB, 0, stream>>>(xr, Wrb, r32, nullptr, nullptr, BT, DD, DD);
  gemm_bt<0><<<ggrid(BT, DD), TPB, 0, stream>>>(xk, Wkb, k32, nullptr, nullptr, BT, DD, DD);
  gemm_bt<0><<<ggrid(BT, DD), TPB, 0, stream>>>(xv, Wvb, v32, nullptr, nullptr, BT, DD, DD);
  gemm_bt<0><<<ggrid(BT, DD), TPB, 0, stream>>>(xg, Wgb, gpre, nullptr, nullptr, BT, DD, DD);

  // decay path: w = -exp(time_decay + tanh(xw@dw1)@dw2)
  gemm_bt<3><<<ggrid(BT, 64), TPB, 0, stream>>>(xw, dw1T, nullptr, tmpd, nullptr, BT, 64, DD);
  gemm_bt<4><<<ggrid(BT, DD), TPB, 0, stream>>>(tmpd, dw2T, wneg, nullptr, tdec, BT, DD, 64);

  // cumsum + shifts, exp factors, v transpose
  cumsum_k<<<dim3(DD / TPB, BBn), TPB, 0, stream>>>(wneg, cs, shF, shB);
  factors_k<<<gElem, TPB, 0, stream>>>(r32, k32, cs, wneg, shF, shB, rff, kff, rbb, kbb);
  vtrans_k<<<gElem, TPB, 0, stream>>>(v32, vT);

  // attention: one wave per workgroup, scalar control flow
  lion_attn_k<<<dim3(TT / 16, BBn * HH), 32, 0, stream>>>(rff, kff, rbb, kbb, vT, yb);

  // groupnorm + gate, then output projection
  gn_gate_k<<<(BT * HH) / 8, TPB, 0, stream>>>(yb, gpre, lnw, lnb, afin);
  gemm_bt<0><<<ggrid(BT, DD), TPB, 0, stream>>>(afin, Wob, out, nullptr, nullptr, BT, DD, DD);
}
